// RefinedGenerator_12558484373697
// MI455X (gfx1250) — matrix operand order repaired and checked by reading, not compiled, not run on hardware
//
#include <hip/hip_runtime.h>
#include <hip/hip_bf16.h>

// ---------------------------------------------------------------------------
// CDNA5 (gfx1250) implementation: f16-input / f32-accumulate WMMA GEMMs for
// all matrix-heavy stages, register-blocked 16x64 strips per wave (1 A-frag ->
// 4 v_wmma), plain VALU kernels for the cheap glue stages.
// ---------------------------------------------------------------------------

typedef _Float16 h16;
typedef __attribute__((ext_vector_type(16))) _Float16 v16h;
typedef __attribute__((ext_vector_type(8)))  float    v8f;

__device__ __forceinline__ v8f wmma16(v16h a, v16h b, v8f c) {
  // v_wmma_f32_16x16x32_f16: D = A(16x32 f16) * B(32x16 f16) + C(16x16 f32)
  return __builtin_amdgcn_wmma_f32_16x16x32_f16(false, a, false, b, (short)0, c,
                                                false, false);
}

// A-fragment (16x32, f16) loaded from row-major f32 matrix, per ISA layout:
// lane(0-15): row m0+lane, K = k0+[0..7] & k0+16+[0..7]
// lane(16-31): row m0+(lane-16), K = k0+8+[0..7] & k0+24+[0..7]
__device__ __forceinline__ v16h load_a_f32(const float* __restrict__ A, int lda,
                                           int M, int Kd, int m0, int k0) {
  int lane = threadIdx.x & 31;
  int m = m0 + (lane & 15);
  int kb = k0 + ((lane >> 4) << 3);
  const float* row = A + (long)m * lda;
  bool mok = m < M;
  v16h a;
#pragma unroll
  for (int i = 0; i < 8; ++i) {
    int k1 = kb + i, k2 = kb + 16 + i;
    a[i]     = (_Float16)((mok && k1 < Kd) ? row[k1] : 0.f);
    a[i + 8] = (_Float16)((mok && k2 < Kd) ? row[k2] : 0.f);
  }
  return a;
}

// B-fragment (32x16, f16) from pre-transposed f16 weights WT[N][ldb] (K-major
// per output column): lane holds column n0+(lane&15), K = k0+8*(lane>>4)+[0..7]
// & k0+16+8*(lane>>4)+[0..7]
// ldb is always a multiple of 16 halves -> 16B-aligned 8-half vector loads.
__device__ __forceinline__ v16h load_b_h16(const h16* __restrict__ BT, int ldb,
                                           int n0, int k0) {
  typedef __attribute__((ext_vector_type(8))) _Float16 v8h_t;
  int lane = threadIdx.x & 31;
  int n = n0 + (lane & 15);
  int kb = k0 + ((lane >> 4) << 3);
  const h16* p = BT + (long)n * ldb + kb;
  v8h_t lo = *(const v8h_t*)p;
  v8h_t hi = *(const v8h_t*)(p + 16);
  return __builtin_shufflevector(lo, hi, 0, 1, 2, 3, 4, 5, 6, 7, 8, 9, 10, 11, 12, 13, 14, 15);
}

__device__ __forceinline__ float sigf(float x) { return 1.f / (1.f + __expf(-x)); }

// ---------------------------------------------------------------------------
// Generic WMMA GEMM:  C[M,N] = act(A[M,Kd] * WT^T + bias (+ res))
// lda = ldb = Kd, ldc = N. Requires N % 64 == 0. 8 waves / block, each wave
// computes a 16x64 strip: one A fragment feeds 4 v_wmma per k-step.
// ---------------------------------------------------------------------------
template <int ACT /*0 none,1 relu*/, int RES /*residual add*/>
__global__ void __launch_bounds__(256) gemm_k(const float* __restrict__ A,
                                              const h16* __restrict__ BT,
                                              const float* __restrict__ bias,
                                              const float* __restrict__ res,
                                              float* __restrict__ C,
                                              int M, int N, int Kd) {
  int nt4 = N >> 6;
  int mt = (M + 15) >> 4;
  int ntiles = mt * nt4;
  int tile = blockIdx.x * 8 + (threadIdx.x >> 5);
  if (tile >= ntiles) return;
  int m0 = (tile / nt4) << 4, n0 = (tile % nt4) << 6;
  v8f cc[4] = {};
  for (int k0 = 0; k0 < Kd; k0 += 32) {
    v16h a = load_a_f32(A, Kd, M, Kd, m0, k0);
#pragma unroll
    for (int j = 0; j < 4; ++j)
      cc[j] = wmma16(a, load_b_h16(BT, Kd, n0 + 16 * j, k0), cc[j]);
  }
  int lane = threadIdx.x & 31;
  int mb = m0 + ((lane >> 4) << 3);
#pragma unroll
  for (int j = 0; j < 4; ++j) {
    int n = n0 + 16 * j + (lane & 15);
    float bv = bias ? bias[n] : 0.f;
#pragma unroll
    for (int r = 0; r < 8; ++r) {
      int m = mb + r;
      if (m < M) {
        float v = cc[j][r] + bv;
        if (ACT == 1) v = v > 0.f ? v : 0.f;
        if (RES) v += res[(long)m * N + n];
        C[(long)m * N + n] = v;
      }
    }
  }
}

// Weight convert+transpose: W f32 [K][N] row-major -> WT f16 [N][Kp], zero-pad.
__global__ void wtrans_k(const float* __restrict__ W, h16* __restrict__ WT,
                         int K, int N, int Kp) {
  long i = (long)blockIdx.x * blockDim.x + threadIdx.x;
  long tot = (long)N * Kp;
  if (i >= tot) return;
  int n = (int)(i / Kp), k = (int)(i % Kp);
  WT[i] = (h16)((k < K) ? W[(long)k * N + n] : 0.f);
}

// ---------------------------------------------------------------------------
// Block-cooperative GEMM helper (GRU scan). Requires N % 64 == 0.
// ---------------------------------------------------------------------------
__device__ void blk_gemm(const float* A, const h16* BT, const float* bias,
                         float* C, int M, int N, int Kd) {
  int nt4 = N >> 6;
  int mt = (M + 15) >> 4;
  int ntiles = mt * nt4;
  int nw = blockDim.x >> 5;
  int lane = threadIdx.x & 31;
  for (int tile = threadIdx.x >> 5; tile < ntiles; tile += nw) {
    int m0 = (tile / nt4) << 4, n0 = (tile % nt4) << 6;
    v8f cc[4] = {};
    for (int k0 = 0; k0 < Kd; k0 += 32) {
      v16h a = load_a_f32(A, Kd, M, Kd, m0, k0);
#pragma unroll
      for (int j = 0; j < 4; ++j)
        cc[j] = wmma16(a, load_b_h16(BT, Kd, n0 + 16 * j, k0), cc[j]);
    }
    int mb = m0 + ((lane >> 4) << 3);
#pragma unroll
    for (int j = 0; j < 4; ++j) {
      int n = n0 + 16 * j + (lane & 15);
      float bv = bias ? bias[n] : 0.f;
#pragma unroll
      for (int r = 0; r < 8; ++r) {
        int m = mb + r;
        if (m < M) C[m * N + n] = cc[j][r] + bv;
      }
    }
  }
}

// ---------------------------------------------------------------------------
// Stage 1: autoregressive stacked GRU (single block, 512 sequential steps).
// x,h live in global scratch (CU-mode: __syncthreads orders them in-block);
// gate buffers in 48KB LDS.
// ---------------------------------------------------------------------------
__global__ void __launch_bounds__(256) gru_scan_k(
    const float* __restrict__ latent, const float* __restrict__ noise,
    const h16* __restrict__ wiT, const h16* __restrict__ whT,
    const float* bi0, const float* bi1, const float* bi2,
    const float* bh0, const float* bh1, const float* bh2,
    const h16* __restrict__ meanT, const float* __restrict__ meanB,
    const h16* __restrict__ lvT, const float* __restrict__ lvB,
    float* __restrict__ xh, float* __restrict__ xseq) {
  __shared__ float gsh[12288];           // gi[32][192] | gh[32][192]
  float* gi = gsh;
  float* gh = gsh + 6144;
  float* x = xh;                          // [32][64]
  float* h = xh + 2048;                   // 3 x [32][64]
  const float* biA[3] = {bi0, bi1, bi2};
  const float* bhA[3] = {bh0, bh1, bh2};
  for (int i = threadIdx.x; i < 2048; i += 256) {
    x[i] = latent[i];
    h[i] = 0.f; h[2048 + i] = 0.f; h[4096 + i] = 0.f;
  }
  __syncthreads();
  for (int t = 0; t < 512; ++t) {
    for (int c = 0; c < 3; ++c) {
      blk_gemm(x, wiT + c * 12288, biA[c], gi, 32, 192, 64);
      blk_gemm(h + c * 2048, whT + c * 12288, bhA[c], gh, 32, 192, 64);
      __syncthreads();
      for (int i = threadIdx.x; i < 2048; i += 256) {
        int b = i >> 6, j = i & 63;
        float r = sigf(gi[b * 192 + j] + gh[b * 192 + j]);
        float z = sigf(gi[b * 192 + 64 + j] + gh[b * 192 + 64 + j]);
        float n = tanhf(gi[b * 192 + 128 + j] + r * gh[b * 192 + 128 + j]);
        float h2 = (1.f - z) * n + z * h[c * 2048 + i];
        h[c * 2048 + i] = h2;
        x[i] = h2;
      }
      __syncthreads();
    }
    blk_gemm(x, meanT, meanB, gi, 32, 64, 64);          // mean -> gi[0:2048)
    blk_gemm(x, lvT, lvB, gi + 2048, 32, 64, 64);       // logvar -> gi[2048:)
    __syncthreads();
    for (int i = threadIdx.x; i < 2048; i += 256) {
      int b = i >> 6, j = i & 63;
      float o = noise[((long)(b << 9) + t) * 64 + j] * __expf(gi[2048 + i] * 0.5f)
                + gi[i];
      x[i] = o;
      xseq[((long)(b << 9) + t) * 64 + j] = o;
    }
    __syncthreads();
  }
}

// ---------------------------------------------------------------------------
// Stage 2: LSTM recurrence. 25 persistent blocks; block b owns hidden units
// [32b,32b+32) and their 4 gate column slices; grid barrier each timestep.
// Wave w = (gate, m-tile): one A fragment -> 2 v_wmma per k-step.
// ---------------------------------------------------------------------------
__device__ void gsync(unsigned* cnt, unsigned target) {
  __threadfence();
  __syncthreads();
  if (threadIdx.x == 0) {
    atomicAdd(cnt, 1u);
    while (__hip_atomic_load(cnt, __ATOMIC_ACQUIRE, __HIP_MEMORY_SCOPE_AGENT) <
           target)
      __builtin_amdgcn_s_sleep(1);
  }
  __syncthreads();
}

__global__ void __launch_bounds__(256) lstm_scan_k(
    const float* __restrict__ xi /*[B,L,3200] with bias*/,
    const h16* __restrict__ whhT /*[3200][800]*/,
    float* __restrict__ hA, float* __restrict__ hB, float* __restrict__ cbuf,
    float* __restrict__ hseq /*[B,L,1600]*/, int dirOff, int reverse,
    unsigned* __restrict__ cnt) {
  __shared__ float gsh[32 * 128];         // [batch][gate*32+j]
  int hu0 = blockIdx.x * 32;
  int w = threadIdx.x >> 5, lane = threadIdx.x & 31;
  int gate = w >> 1;
  int m0 = (w & 1) << 4;                  // m-tile of this wave
  int n0 = gate * 800 + hu0;              // 32-wide gate slice
  for (int i = threadIdx.x; i < 1024; i += 256) {
    int b = i >> 5, j = i & 31;
    hA[b * 800 + hu0 + j] = 0.f;
    hB[b * 800 + hu0 + j] = 0.f;
    cbuf[b * 800 + hu0 + j] = 0.f;
  }
  unsigned gen = gridDim.x;
  gsync(cnt, gen);
  for (int step = 0; step < 512; ++step) {
    int t = reverse ? (511 - step) : step;
    const float* hin = (step & 1) ? hB : hA;
    float* hout = (step & 1) ? hA : hB;
    v8f c0 = {}, c1 = {};
    for (int k0 = 0; k0 < 800; k0 += 32) {
      v16h a = load_a_f32(hin, 800, 32, 800, m0, k0);
      c0 = wmma16(a, load_b_h16(whhT, 800, n0, k0), c0);
      c1 = wmma16(a, load_b_h16(whhT, 800, n0 + 16, k0), c1);
    }
    {
      int nl0 = gate * 32 + (lane & 15);
      int mb = m0 + ((lane >> 4) << 3);
#pragma unroll
      for (int r = 0; r < 8; ++r) {
        gsh[(mb + r) * 128 + nl0] = c0[r];
        gsh[(mb + r) * 128 + nl0 + 16] = c1[r];
      }
    }
    __syncthreads();
    for (int i = threadIdx.x; i < 1024; i += 256) {
      int b = i >> 5, j = i & 31;
      int hu = hu0 + j;
      long xo = ((long)(b << 9) + t) * 3200;
      float ig = gsh[b * 128 + j]      + xi[xo + hu];
      float fg = gsh[b * 128 + 32 + j] + xi[xo + 800 + hu];
      float gg = gsh[b * 128 + 64 + j] + xi[xo + 1600 + hu];
      float og = gsh[b * 128 + 96 + j] + xi[xo + 2400 + hu];
      float c2 = sigf(fg) * cbuf[b * 800 + hu] + sigf(ig) * tanhf(gg);
      float h2 = sigf(og) * tanhf(c2);
      cbuf[b * 800 + hu] = c2;
      hout[b * 800 + hu] = h2;
      hseq[((long)(b << 9) + t) * 1600 + dirOff + hu] = h2;
    }
    gen += gridDim.x;
    gsync(cnt, gen);
  }
}

// ---------------------------------------------------------------------------
// Stage 3: angles + sin/cos features, positions, frames, kNN, rel features.
// ---------------------------------------------------------------------------
__global__ void angle_k(const float* __restrict__ hseq2,
                        const float* __restrict__ w, const float* __restrict__ b,
                        float* __restrict__ angles, float* __restrict__ afeat) {
  int r = blockIdx.x * blockDim.x + threadIdx.x;
  if (r >= 16384) return;
  const float* hp = hseq2 + (long)r * 1600;
  float a0 = b[0], a1 = b[1], a2 = b[2];
  for (int k = 0; k < 1600; ++k) {
    float h = hp[k];
    a0 += h * w[k * 3]; a1 += h * w[k * 3 + 1]; a2 += h * w[k * 3 + 2];
  }
  angles[r * 3] = a0; angles[r * 3 + 1] = a1; angles[r * 3 + 2] = a2;
  float* af = afeat + (long)r * 6;
  af[0] = sinf(a0); af[1] = sinf(a1); af[2] = sinf(a2);
  af[3] = cosf(a0); af[4] = cosf(a1); af[5] = cosf(a2);
}

__global__ void pos_k(const float* __restrict__ angles, float* __restrict__ pos) {
  int b = blockIdx.x, ax = threadIdx.x;
  if (ax >= 3) return;
  float acc = 0.f;
  for (int l = 0; l < 512; ++l) {
    int r = (b << 9) + l;
    float phi = angles[r * 3], psi = angles[r * 3 + 1];
    float s = (ax == 0) ? cosf(phi)
                        : ((ax == 1) ? sinf(phi) * cosf(psi) : sinf(phi) * sinf(psi));
    acc += 3.8f * s;
    pos[r * 3 + ax] = acc;
  }
}

__global__ void ors_k(const float* __restrict__ pos, float* __restrict__ ors) {
  int r = blockIdx.x * blockDim.x + threadIdx.x;
  if (r >= 16384) return;
  int l = r & 511, base = r - l;
  int lu = (l < 511) ? l : 510;
  int lv = (l > 0) ? l - 1 : 0;
  float ux = pos[(base + lu + 1) * 3]     - pos[(base + lu) * 3];
  float uy = pos[(base + lu + 1) * 3 + 1] - pos[(base + lu) * 3 + 1];
  float uz = pos[(base + lu + 1) * 3 + 2] - pos[(base + lu) * 3 + 2];
  float vx = pos[(base + lv + 1) * 3]     - pos[(base + lv) * 3];
  float vy = pos[(base + lv + 1) * 3 + 1] - pos[(base + lv) * 3 + 1];
  float vz = pos[(base + lv + 1) * 3 + 2] - pos[(base + lv) * 3 + 2];
  float un = sqrtf(ux * ux + uy * uy + uz * uz) + 1e-6f; ux /= un; uy /= un; uz /= un;
  float vn = sqrtf(vx * vx + vy * vy + vz * vz) + 1e-6f; vx /= vn; vy /= vn; vz /= vn;
  float cx = uy * vz - uz * vy + 1e-6f;
  float cy = uz * vx - ux * vz + 1e-6f;
  float cz = ux * vy - uy * vx + 1e-6f;
  float cn = sqrtf(cx * cx + cy * cy + cz * cz) + 1e-6f; cx /= cn; cy /= cn; cz /= cn;
  float bx = cy * uz - cz * uy, by = cz * ux - cx * uz, bz = cx * uy - cy * ux;
  float* o = ors + (long)r * 9;
  o[0] = ux; o[1] = uy; o[2] = uz; o[3] = cx; o[4] = cy; o[5] = cz;
  o[6] = bx; o[7] = by; o[8] = bz;
}

__global__ void knn_k(const float* __restrict__ pos, int* __restrict__ nbr) {
  int r = blockIdx.x * blockDim.x + threadIdx.x;
  if (r >= 16384) return;
  int base = r & ~511;
  float px = pos[r * 3], py = pos[r * 3 + 1], pz = pos[r * 3 + 2];
  float bd[15]; int bi[15];
#pragma unroll
  for (int i = 0; i < 15; ++i) { bd[i] = -3.4e38f; bi[i] = 0; }
  for (int j = 0; j < 512; ++j) {
    float dx = px - pos[(base + j) * 3];
    float dy = py - pos[(base + j) * 3 + 1];
    float dz = pz - pos[(base + j) * 3 + 2];
    float c = -sqrtf(dx * dx + dy * dy + dz * dz + 1e-12f);
    if (c > bd[14]) {
      int p = 14;
      while (p > 0 && c > bd[p - 1]) { bd[p] = bd[p - 1]; bi[p] = bi[p - 1]; --p; }
      bd[p] = c; bi[p] = j;
    }
  }
  for (int i = 0; i < 15; ++i) nbr[r * 15 + i] = bi[i];
}

__global__ void rel_k(const float* __restrict__ pos, const float* __restrict__ ors,
                      const int* __restrict__ nbr, float* __restrict__ reldata) {
  int idx = blockIdx.x * blockDim.x + threadIdx.x;
  if (idx >= 16384 * 15) return;
  int r = idx / 15;
  int rj = (r & ~511) + nbr[idx];
  float rx = pos[r * 3]     - pos[rj * 3]     + 1e-6f;
  float ry = pos[r * 3 + 1] - pos[rj * 3 + 1] + 1e-6f;
  float rz = pos[r * 3 + 2] - pos[rj * 3 + 2] + 1e-6f;
  float dist = sqrtf(rx * rx + ry * ry + rz * rz);
  float* o = reldata + (long)idx * 76;
  const float invsig = 63.f / 20.f;
  for (int c = 0; c < 64; ++c) {
    float t = (dist - 20.f * c / 63.f) * invsig;
    o[c] = __expf(-t * t);
  }
  o[64] = rx; o[65] = ry; o[66] = rz;
  for (int i = 0; i < 9; ++i) o[67 + i] = ors[(long)r * 9 + i] - ors[(long)rj * 9 + i];
}

// ---------------------------------------------------------------------------
// Stage 4: local conv features (conv_pre + InstanceNorm + dilated conv_blk).
// ---------------------------------------------------------------------------
__global__ void convpre_k(const float* __restrict__ afeat, const float* __restrict__ w,
                          const float* __restrict__ bias, float* __restrict__ out1) {
  int idx = blockIdx.x * blockDim.x + threadIdx.x;
  if (idx >= 32 * 128 * 512) return;
  int l = idx & 511, co = (idx >> 9) & 127, b = idx >> 16;
  float acc = bias[co];
  for (int ci = 0; ci < 6; ++ci)
    for (int t = 0; t < 3; ++t) {
      int ll = l + t - 1;
      if (ll >= 0 && ll < 512)
        acc += w[(co * 6 + ci) * 3 + t] * afeat[((long)(b << 9) + ll) * 6 + ci];
    }
  out1[idx] = acc > 0.f ? acc : (__expf(acc) - 1.f);
}

__global__ void inorm_k(const float* __restrict__ out1, float* __restrict__ normed) {
  __shared__ float s1[256], s2[256];
  const float* x = out1 + (long)blockIdx.x * 512;
  float a = 0.f, b = 0.f;
  for (int l = threadIdx.x; l < 512; l += 256) { float v = x[l]; a += v; b += v * v; }
  s1[threadIdx.x] = a; s2[threadIdx.x] = b;
  __syncthreads();
  for (int s = 128; s > 0; s >>= 1) {
    if (threadIdx.x < s) { s1[threadIdx.x] += s1[threadIdx.x + s]; s2[threadIdx.x] += s2[threadIdx.x + s]; }
    __syncthreads();
  }
  float mean = s1[0] / 512.f;
  float var = s2[0] / 512.f - mean * mean;
  float inv = 1.f / sqrtf(var + 1e-5f);
  for (int l = threadIdx.x; l < 512; l += 256)
    normed[(long)blockIdx.x * 512 + l] = (x[l] - mean) * inv;
}

__global__ void convblk_k(const float* __restrict__ normed, const float* __restrict__ out1,
                          const float* __restrict__ w, const float* __restrict__ bias,
                          float* __restrict__ feats) {
  int idx = blockIdx.x * blockDim.x + threadIdx.x;
  if (idx >= 32 * 128 * 512) return;
  int l = idx & 511, co = (idx >> 9) & 127, b = idx >> 16;
  float acc = bias[co];
  const float* nb = normed + (long)(b * 128) * 512;
  for (int ci = 0; ci < 128; ++ci) {
    const float* wr = w + (co * 128 + ci) * 3;
    const float* xr = nb + ci * 512;
    if (l >= 2) acc += wr[0] * xr[l - 2];
    acc += wr[1] * xr[l];
    if (l + 2 < 512) acc += wr[2] * xr[l + 2];
  }
  acc += out1[idx];
  float v = acc > 0.f ? acc : (__expf(acc) - 1.f);
  feats[((long)(b << 9) + l) * 128 + co] = v;
}

// ---------------------------------------------------------------------------
// Stage 5: transformer blocks. k/v projections are fused-gather WMMA GEMMs
// over M = B*L*K rows of kin = [f[nbr] (128) | reldata (76) | pad (20)].
// Each wave: 16x64 strip (one gathered A fragment -> 4 v_wmma per k-step).
// ---------------------------------------------------------------------------
__device__ __forceinline__ float kin_elem(const float* frow, const float* rel, int k) {
  if (k < 128) return frow[k];
  k -= 128;
  return (k < 76) ? rel[k] : 0.f;
}

__global__ void __launch_bounds__(256) kv_k(const float* __restrict__ f,
                                            const float* __restrict__ reldata,
                                            const int* __restrict__ nbr,
                                            const h16* __restrict__ BT /*[128][224]*/,
                                            const float* __restrict__ bias,
                                            float* __restrict__ out /*[245760][128]*/) {
  int tile = blockIdx.x * 8 + (threadIdx.x >> 5);   // 15360 m-tiles x 2 strips
  int m0 = (tile >> 1) << 4, n0 = (tile & 1) << 6;
  int lane = threadIdx.x & 31;
  int m = m0 + (lane & 15);
  int rr = m / 15;
  int j = nbr[m];
  const float* frow = f + ((long)(rr & ~511) + j) * 128;
  const float* relrow = reldata + (long)m * 76;
  v8f cc[4] = {};
  for (int k0 = 0; k0 < 224; k0 += 32) {
    int kb = k0 + ((lane >> 4) << 3);
    v16h a;
#pragma unroll
    for (int i = 0; i < 8; ++i) {
      a[i]     = (_Float16)kin_elem(frow, relrow, kb + i);
      a[i + 8] = (_Float16)kin_elem(frow, relrow, kb + 16 + i);
    }
#pragma unroll
    for (int jt = 0; jt < 4; ++jt)
      cc[jt] = wmma16(a, load_b_h16(BT, 224, n0 + 16 * jt, k0), cc[jt]);
  }
  int mb = m0 + ((lane >> 4) << 3);
#pragma unroll
  for (int jt = 0; jt < 4; ++jt) {
    int n = n0 + 16 * jt + (lane & 15);
    float bv = bias[n];
#pragma unroll
    for (int r = 0; r < 8; ++r) out[(long)(mb + r) * 128 + n] = cc[jt][r] + bv;
  }
}

__global__ void attn_k(const float* __restrict__ q, const float* __restrict__ kk,
                       const float* __restrict__ vv, float* __restrict__ ao) {
  int idx = blockIdx.x * blockDim.x + threadIdx.x;   // 16384*8
  if (idx >= 131072) return;
  int r = idx >> 3, h = idx & 7;
  const float* qp = q + (long)r * 128 + h * 16;
  float sc[15], mx = -3.4e38f;
  for (int kn = 0; kn < 15; ++kn) {
    const float* kp = kk + ((long)r * 15 + kn) * 128 + h * 16;
    float d = 0.f;
#pragma unroll
    for (int i = 0; i < 16; ++i) d += qp[i] * kp[i];
    d *= 0.25f;                                       // 1/sqrt(DH)
    sc[kn] = d; mx = fmaxf(mx, d);
  }
  float s = 0.f;
  for (int kn = 0; kn < 15; ++kn) { sc[kn] = __expf(sc[kn] - mx); s += sc[kn]; }
  float inv = 1.f / s;
  float acc[16];
#pragma unroll
  for (int i = 0; i < 16; ++i) acc[i] = 0.f;
  for (int kn = 0; kn < 15; ++kn) {
    float wv = sc[kn] * inv;
    const float* vp = vv + ((long)r * 15 + kn) * 128 + h * 16;
#pragma unroll
    for (int i = 0; i < 16; ++i) acc[i] += wv * vp[i];
  }
  float* op = ao + (long)r * 128 + h * 16;
#pragma unroll
  for (int i = 0; i < 16; ++i) op[i] = acc[i];
}

__global__ void corr_k(const float* __restrict__ f, const float* __restrict__ w,
                       const float* __restrict__ b, const float* __restrict__ angles,
                       float* __restrict__ out) {
  int r = blockIdx.x * blockDim.x + threadIdx.x;
  if (r >= 16384) return;
  const float* fp = f + (long)r * 128;
  float a0 = b[0], a1 = b[1], a2 = b[2];
  for (int k = 0; k < 128; ++k) {
    float v = fp[k];
    a0 += v * w[k * 3]; a1 += v * w[k * 3 + 1]; a2 += v * w[k * 3 + 2];
  }
  out[r * 3]     = angles[r * 3]     + a0;
  out[r * 3 + 1] = angles[r * 3 + 1] + a1;
  out[r * 3 + 2] = angles[r * 3 + 2] + a2;
}

// ---------------------------------------------------------------------------
// Host orchestration.
// Input leaf order (pytree, dict keys sorted; {w,b} -> b,w):
//  0 latent, 1 noise, 2/3 angle_lookup b/w, 4/5 ar_logvar b/w, 6/7 ar_mean b/w,
//  8..35 blocks[2] x {k,m1,m2,m3,o,q,v} x {b,w},
//  36/37 conv_blk b/w, 38/39 conv_pre b/w, 40/41 corr b/w, 42/43 enc_pre b/w,
//  44..55 gru[3] x {bh,bi,wh,wi},
//  56..67 lstm[2] x {bwd,fwd} x {b,whh,wih}.
// ---------------------------------------------------------------------------
extern "C" void kernel_launch(void* const* d_in, const int* in_sizes, int n_in,
                              void* d_out, int out_size, void* d_ws, size_t ws_size,
                              hipStream_t stream) {
  (void)in_sizes; (void)n_in; (void)out_size; (void)ws_size;
  auto F = [&](int i) { return (const float*)d_in[i]; };
  char* p = (char*)d_ws;
  auto alloc = [&](size_t bytes) {
    void* r = (void*)p;
    p += (bytes + 255) & ~(size_t)255;
    return r;
  };
  // ---- f16 transposed weights ----
  h16* gruWiT = (h16*)alloc(3 * 12288 * 2);
  h16* gruWhT = (h16*)alloc(3 * 12288 * 2);
  h16* meanT  = (h16*)alloc(4096 * 2);
  h16* lvT    = (h16*)alloc(4096 * 2);
  h16* encT   = (h16*)alloc(16384 * 2);
  h16 *qT[2], *kT[2], *vT[2], *oT[2], *m1T[2], *m2T[2], *m3T[2];
  for (int b = 0; b < 2; ++b) {
    qT[b]  = (h16*)alloc(16384 * 2);
    kT[b]  = (h16*)alloc(28672 * 2);
    vT[b]  = (h16*)alloc(28672 * 2);
    oT[b]  = (h16*)alloc(16384 * 2);
    m1T[b] = (h16*)alloc(16384 * 2);
    m2T[b] = (h16*)alloc(16384 * 2);
    m3T[b] = (h16*)alloc(16384 * 2);
  }
  h16* wihT_l0f = (h16*)alloc(204800 * 2);
  h16* wihT_l0b = (h16*)alloc(204800 * 2);
  h16* wihT_l1f = (h16*)alloc((size_t)5120000 * 2);
  h16* wihT_l1b = (h16*)alloc((size_t)5120000 * 2);
  h16* whhT_l0f = (h16*)alloc((size_t)2560000 * 2);
  h16* whhT_l0b = (h16*)alloc((size_t)2560000 * 2);
  h16* whhT_l1f = (h16*)alloc((size_t)2560000 * 2);
  h16* whhT_l1b = (h16*)alloc((size_t)2560000 * 2);
  // ---- f32 activations ----
  float* gruXH  = (float*)alloc(4 * 2048 * 4);
  float* xseq   = (float*)alloc((size_t)1048576 * 4);
  float* hseq1  = (float*)alloc((size_t)26214400 * 4);
  float* hseq2  = (float*)alloc((size_t)26214400 * 4);
  float* big    = (float*)alloc((size_t)62914560 * 4);   // xi  OR  kk|vv
  float* xi = big;
  float* kk = big;
  float* vv = big + (size_t)31457280;
  float* hA   = (float*)alloc(25600 * 4);
  float* hB   = (float*)alloc(25600 * 4);
  float* cbuf = (float*)alloc(25600 * 4);
  float* angles = (float*)alloc(49152 * 4);
  float* afeat  = (float*)alloc(98304 * 4);
  float* pos    = (float*)alloc(49152 * 4);
  float* ors    = (float*)alloc(147456 * 4);
  int*   nbr    = (int*)alloc(245760 * 4);
  float* reldata = (float*)alloc((size_t)18677760 * 4);
  float* out1   = (float*)alloc((size_t)2097152 * 4);
  float* normed = (float*)alloc((size_t)2097152 * 4);
  float* feats  = (float*)alloc((size_t)2097152 * 4);
  float* fbuf   = (float*)alloc((size_t)2097152 * 4);
  float* qbuf   = (float*)alloc((size_t)2097152 * 4);
  float* aobuf  = (float*)alloc((size_t)2097152 * 4);
  float* t1     = (float*)alloc((size_t)2097152 * 4);
  float* t2     = (float*)alloc((size_t)2097152 * 4);
  unsigned* cnt = (unsigned*)alloc(64);

  hipMemsetAsync(cnt, 0, 64, stream);

  auto wt = [&](const float* W, h16* WT, int K, int N, int Kp) {
    long tot = (long)N * Kp;
    wtrans_k<<<(unsigned)((tot + 255) / 256), 256, 0, stream>>>(W, WT, K, N, Kp);
  };
  for (int c = 0; c < 3; ++c) {
    wt(F(44 + 4 * c + 3), gruWiT + c * 12288, 64, 192, 64);   // wi
    wt(F(44 + 4 * c + 2), gruWhT + c * 12288, 64, 192, 64);   // wh
  }
  wt(F(7), meanT, 64, 64, 64);
  wt(F(5), lvT, 64, 64, 64);
  wt(F(43), encT, 128, 128, 128);
  for (int b = 0; b < 2; ++b) {
    int base = 8 + 14 * b;
    wt(F(base + 1),  kT[b], 204, 128, 224);
    wt(F(base + 3),  m1T[b], 128, 128, 128);
    wt(F(base + 5),  m2T[b], 128, 128, 128);
    wt(F(base + 7),  m3T[b], 128, 128, 128);
    wt(F(base + 9),  oT[b], 128, 128, 128);
    wt(F(base + 11), qT[b], 128, 128, 128);
    wt(F(base + 13), vT[b], 204, 128, 224);
  }
  wt(F(57), whhT_l0b, 800, 3200, 800);
  wt(F(58), wihT_l0b, 64, 3200, 64);
  wt(F(60), whhT_l0f, 800, 3200, 800);
  wt(F(61), wihT_l0f, 64, 3200, 64);
  wt(F(63), whhT_l1b, 800, 3200, 800);
  wt(F(64), wihT_l1b, 1600, 3200, 1600);
  wt(F(66), whhT_l1f, 800, 3200, 800);
  wt(F(67), wihT_l1f, 1600, 3200, 1600);

  // Stage 1: AR GRU
  gru_scan_k<<<1, 256, 0, stream>>>(F(0), F(1), gruWiT, gruWhT,
                                    F(45), F(49), F(53), F(44), F(48), F(52),
                                    meanT, F(6), lvT, F(4), gruXH, xseq);

  auto gemm0 = [&](const float* A, const h16* BT, const float* bias, float* C,
                   int M, int N, int K) {
    int tiles = ((M + 15) / 16) * (N / 64);
    gemm_k<0, 0><<<(tiles + 7) / 8, 256, 0, stream>>>(A, BT, bias, nullptr, C, M, N, K);
  };
  auto gemmRelu = [&](const float* A, const h16* BT, const float* bias, float* C,
                      int M, int N, int K) {
    int tiles = ((M + 15) / 16) * (N / 64);
    gemm_k<1, 0><<<(tiles + 7) / 8, 256, 0, stream>>>(A, BT, bias, nullptr, C, M, N, K);
  };
  auto gemmRes = [&](const float* A, const h16* BT, const float* bias,
                     const float* res, float* C, int M, int N, int K) {
    int tiles = ((M + 15) / 16) * (N / 64);
    gemm_k<0, 1><<<(tiles + 7) / 8, 256, 0, stream>>>(A, BT, bias, res, C, M, N, K);
  };

  // Stage 2: BiLSTM (concat order: fwd at offset 0, bwd at offset 800)
  gemm0(xseq, wihT_l0f, F(59), xi, 16384, 3200, 64);
  lstm_scan_k<<<25, 256, 0, stream>>>(xi, whhT_l0f, hA, hB, cbuf, hseq1, 0, 0, cnt + 0);
  gemm0(xseq, wihT_l0b, F(56), xi, 16384, 3200, 64);
  lstm_scan_k<<<25, 256, 0, stream>>>(xi, whhT_l0b, hA, hB, cbuf, hseq1, 800, 1, cnt + 1);
  gemm0(hseq1, wihT_l1f, F(65), xi, 16384, 3200, 1600);
  lstm_scan_k<<<25, 256, 0, stream>>>(xi, whhT_l1f, hA, hB, cbuf, hseq2, 0, 0, cnt + 2);
  gemm0(hseq1, wihT_l1b, F(62), xi, 16384, 3200, 1600);
  lstm_scan_k<<<25, 256, 0, stream>>>(xi, whhT_l1b, hA, hB, cbuf, hseq2, 800, 1, cnt + 3);

  // Stage 3: angles, geometry, kNN, rel features
  angle_k<<<64, 256, 0, stream>>>(hseq2, F(3), F(2), angles, afeat);
  pos_k<<<32, 32, 0, stream>>>(angles, pos);
  ors_k<<<64, 256, 0, stream>>>(pos, ors);
  knn_k<<<64, 256, 0, stream>>>(pos, nbr);
  rel_k<<<960, 256, 0, stream>>>(pos, ors, nbr, reldata);

  // Stage 4: conv feature stack
  convpre_k<<<8192, 256, 0, stream>>>(afeat, F(39), F(38), out1);
  inorm_k<<<4096, 256, 0, stream>>>(out1, normed);
  convblk_k<<<8192, 256, 0, stream>>>(normed, out1, F(37), F(36), feats);

  // Stage 5: transformer encoder
  gemm0(feats, encT, F(42), fbuf, 16384, 128, 128);
  for (int b = 0; b < 2; ++b) {
    int base = 8 + 14 * b;
    gemm0(fbuf, qT[b], F(base + 10), qbuf, 16384, 128, 128);
    kv_k<<<3840, 256, 0, stream>>>(fbuf, reldata, nbr, kT[b], F(base + 0), kk);
    kv_k<<<3840, 256, 0, stream>>>(fbuf, reldata, nbr, vT[b], F(base + 12), vv);
    attn_k<<<512, 256, 0, stream>>>(qbuf, kk, vv, aobuf);
    gemmRes(aobuf, oT[b], F(base + 8), fbuf, fbuf, 16384, 128, 128);
    gemmRelu(fbuf, m1T[b], F(base + 2), t1, 16384, 128, 128);
    gemmRelu(t1, m2T[b], F(base + 4), t2, 16384, 128, 128);
    gemmRes(t2, m3T[b], F(base + 6), fbuf, fbuf, 16384, 128, 128);
  }

  // Stage 6: output correction
  corr_k<<<64, 256, 0, stream>>>(fbuf, F(41), F(40), angles, (float*)d_out);
}
